// CustomLinearLayer_52338471469274
// MI455X (gfx1250) — matrix operand order repaired and checked by reading, not compiled, not run on hardware
//
#include <hip/hip_runtime.h>
#include <hip/hip_bf16.h>

typedef __bf16 bf16_t;
typedef __attribute__((ext_vector_type(16))) __bf16 v16bf;
typedef __attribute__((ext_vector_type(8)))  __bf16 v8bf;
typedef __attribute__((ext_vector_type(8)))  float  v8f;

#define BATCH   4096
#define OUT_DIM 512
#define IN_DIM  2048

#define BM 128            // block tile M
#define BN 128            // block tile N
#define BK 64             // block tile K
#define LDA (BK + 8)      // LDS row stride (bf16 elems), 144 B rows -> conflict-friendly
#define LDB (BK + 8)

// f32 pair -> packed bf16x2 in one dword via the HW packed convert (RNE).
// The clang builtin is not declared for gfx1250, so emit the instruction directly.
__device__ __forceinline__ unsigned pack2(float lo, float hi) {
    unsigned r;
    asm("v_cvt_pk_bf16_f32 %0, %1, %2" : "=v"(r) : "v"(lo), "v"(hi));
    return r;
}

// Stage one 128x64 A tile (f32 -> bf16) and one 128x64 B tile (S*THETA -> bf16).
__device__ __forceinline__ void stage_tiles(bf16_t* __restrict__ Asb,
                                            bf16_t* __restrict__ Bsb,
                                            const float* __restrict__ input,
                                            const float* __restrict__ S,
                                            const float* __restrict__ THETA,
                                            unsigned aIdx, unsigned bIdx,
                                            int rbase, int col4) {
    #pragma unroll
    for (int i = 0; i < 8; ++i) {
        const float4 a = *(const float4*)&input[aIdx + (unsigned)(i * 16 * IN_DIM)];
        *(uint2*)&Asb[(rbase + i * 16) * LDA + col4 * 4] =
            make_uint2(pack2(a.x, a.y), pack2(a.z, a.w));
    }
    #pragma unroll
    for (int i = 0; i < 8; ++i) {
        const unsigned g = bIdx + (unsigned)(i * 16 * IN_DIM);
        const float4 s  = *(const float4*)&S[g];
        const float4 th = *(const float4*)&THETA[g];
        *(uint2*)&Bsb[(rbase + i * 16) * LDB + col4 * 4] =
            make_uint2(pack2(s.x * th.x, s.y * th.y), pack2(s.z * th.z, s.w * th.w));
    }
}

// One K-block of compute: 2 K-steps of 32, 4x2 WMMA tiles per wave
__device__ __forceinline__ void compute_tiles(const bf16_t* __restrict__ Asb,
                                              const bf16_t* __restrict__ Bsb,
                                              v8f (&acc)[4][2],
                                              int wm, int wn, int lane) {
    union Frag { v16bf v; v8bf h[2]; };
    const int half  = lane >> 4;
    const int mrow  = lane & 15;
    const int ncol  = lane & 15;
    const int khalf = lane >> 4;

    #pragma unroll
    for (int kk = 0; kk < BK; kk += 32) {
        // A frags: lane = row (l&15); half picks K {h*8..+7} u {16+h*8..+7}
        Frag a[4];
        #pragma unroll
        for (int tm = 0; tm < 4; ++tm) {
            const int m = wm * 64 + tm * 16 + mrow;
            const bf16_t* p = &Asb[m * LDA + kk + half * 8];
            a[tm].h[0] = *(const v8bf*)p;
            a[tm].h[1] = *(const v8bf*)(p + 16);
        }
        // B frags: lane&15 = N column, lane>>4 picks K {kh*8..+7} u {16+kh*8..+7}
        Frag b[2];
        #pragma unroll
        for (int tn = 0; tn < 2; ++tn) {
            const int n = wn * 32 + tn * 16 + ncol;
            const bf16_t* p = &Bsb[n * LDB + kk + khalf * 8];
            b[tn].h[0] = *(const v8bf*)p;
            b[tn].h[1] = *(const v8bf*)(p + 16);
        }
        #pragma unroll
        for (int tm = 0; tm < 4; ++tm)
            #pragma unroll
            for (int tn = 0; tn < 2; ++tn)
                acc[tm][tn] = __builtin_amdgcn_wmma_f32_16x16x32_bf16(
                    false, a[tm].v, false, b[tn].v,
                    (short)0, acc[tm][tn], false, false);
    }
}

__global__ __launch_bounds__(256)
void masked_linear_wmma_bf16(const float* __restrict__ input,
                             const float* __restrict__ S,
                             const float* __restrict__ THETA,
                             const float* __restrict__ bias,
                             float* __restrict__ out) {
    __shared__ bf16_t As[2][BM * LDA];   // double-buffered A tiles (2 x 18 KB)
    __shared__ bf16_t Bs[2][BN * LDB];   // double-buffered B tiles (2 x 18 KB)

    const int t    = threadIdx.x;
    const int lane = t & 31;
    const int wave = t >> 5;
    const int wm   = wave & 1;    // 2 M slots x 64 rows
    const int wn   = wave >> 1;   // 4 N slots x 32 cols

    const int m0 = blockIdx.y * BM;
    const int n0 = blockIdx.x * BN;

    const int col4  = t & 15;
    const int rbase = t >> 4;

    // 32-bit base element indices for staging
    const unsigned aBase = (unsigned)(m0 + rbase) * IN_DIM + (unsigned)(col4 * 4);
    const unsigned bBase = (unsigned)(n0 + rbase) * IN_DIM + (unsigned)(col4 * 4);

    v8f acc[4][2] = {};           // wave tile: 64(M) x 32(N)

    // prologue: stage K-block 0 into buffer 0
    stage_tiles(As[0], Bs[0], input, S, THETA, aBase, bBase, rbase, col4);

    // main loop: unroll by 2 K-blocks so buffer index is static.
    // One barrier per K-block; staging of block i+1 overlaps compute of block i.
    for (int kb = 0; kb < IN_DIM; kb += 2 * BK) {
        __syncthreads();
        stage_tiles(As[1], Bs[1], input, S, THETA,
                    aBase + (unsigned)(kb + BK), bBase + (unsigned)(kb + BK),
                    rbase, col4);
        compute_tiles(As[0], Bs[0], acc, wm, wn, lane);

        __syncthreads();
        if (kb + 2 * BK < IN_DIM)
            stage_tiles(As[0], Bs[0], input, S, THETA,
                        aBase + (unsigned)(kb + 2 * BK), bBase + (unsigned)(kb + 2 * BK),
                        rbase, col4);
        compute_tiles(As[1], Bs[1], acc, wm, wn, lane);
    }

    // ---- epilogue: C/D layout -> global, add bias ----
    // VGPR v holds row (v + (lane>>4)*8), col = lane&15 within each 16x16 tile
    const int ncol = lane & 15;
    const int rowq = (lane >> 4) * 8;
    #pragma unroll
    for (int tm = 0; tm < 4; ++tm) {
        #pragma unroll
        for (int tn = 0; tn < 2; ++tn) {
            const int gm = m0 + wm * 64 + tm * 16;
            const unsigned gn = (unsigned)(n0 + wn * 32 + tn * 16 + ncol);
            const float bv = bias[gn];
            #pragma unroll
            for (int v = 0; v < 8; ++v) {
                out[(unsigned)(gm + rowq + v) * OUT_DIM + gn] = acc[tm][tn][v] + bv;
            }
        }
    }
}

extern "C" void kernel_launch(void* const* d_in, const int* in_sizes, int n_in,
                              void* d_out, int out_size, void* d_ws, size_t ws_size,
                              hipStream_t stream) {
    (void)in_sizes; (void)n_in; (void)out_size; (void)d_ws; (void)ws_size;
    const float* input = (const float*)d_in[0];
    const float* S     = (const float*)d_in[1];
    const float* THETA = (const float*)d_in[2];
    const float* bias  = (const float*)d_in[3];
    float* out         = (float*)d_out;

    dim3 grid(OUT_DIM / BN, BATCH / BM);   // (4, 32) = 128 blocks
    dim3 block(256);
    masked_linear_wmma_bf16<<<grid, block, 0, stream>>>(input, S, THETA, bias, out);
}
